// Attention_43963285242676
// MI455X (gfx1250) — matrix operand order repaired and checked by reading, not compiled, not run on hardware
//
#include <hip/hip_runtime.h>
#include <hip/hip_bf16.h>

// ---------------------------------------------------------------------------
// MI455X (gfx1250) attention block:
//   cvt(f32->bf16) -> QKV GEMM(+RoPE, scatter) -> flash attention -> O-proj
// Matmuls: v_wmma_f32_16x16x32_bf16 (wave32), f32 accumulation.
// Tile staging: GLOBAL_LOAD_ASYNC_TO_LDS_B128 double-buffered pipeline
// (single barrier per K-step), falling back to sync staging if unavailable.
// ---------------------------------------------------------------------------

typedef __attribute__((ext_vector_type(16))) __bf16 v16bf;
typedef __attribute__((ext_vector_type(8)))  __bf16 bfx8;
typedef __attribute__((ext_vector_type(8)))  float  v8f;
typedef __attribute__((ext_vector_type(4)))  int    v4i;

#define BM   128
#define BN   128
#define BKK  32
#define LDSS 40    // 32 + 8 pad bf16 (80 B rows: 16B aligned, conflict-free A reads)
#define LDK  72    // 64 + 8 pad bf16 (144 B rows)

#define N_HEADS    32
#define N_KV_HEADS 8
#define HEAD_DIM   64
#define LSEQ       2048
#define DMODEL     2048
#define OPSZ       3072
#define SCALE_C    0.125f
#define NEG_BIG    (-1.0e30f)

#if defined(__has_builtin)
#if __has_builtin(__builtin_amdgcn_global_load_async_to_lds_b128) && \
    __has_builtin(__builtin_amdgcn_s_wait_asynccnt)
#define HAVE_ASYNC 1
#endif
#endif
#ifndef HAVE_ASYNC
#define HAVE_ASYNC 0
#endif

#if HAVE_ASYNC
// Builtin prototype (from hipcc diagnostic): (v4i AS1* gsrc, v4i AS3* ldst,
// imm offset, imm cpol). 16-byte global -> LDS async copy, ASYNCcnt-tracked.
typedef __attribute__((address_space(1))) v4i gas_v4i;
typedef __attribute__((address_space(3))) v4i las_v4i;

static __device__ __forceinline__ void async_cp16(const __bf16* g, __bf16* l) {
  __builtin_amdgcn_global_load_async_to_lds_b128((gas_v4i*)g, (las_v4i*)l, 0, 0);
}
#endif

// --- fragment loaders (ISA 7.12.2 layouts) ---------------------------------
// A 16x32 bf16: lane row M=lo; K chunks [hi*8..hi*8+7], [16+hi*8..+7]
static __device__ __forceinline__ v16bf load_frag_a(const __bf16* p) {
  v16bf f;
  ((bfx8*)&f)[0] = *(const bfx8*)(p);
  ((bfx8*)&f)[1] = *(const bfx8*)(p + 16);
  return f;
}
// B 32x16 bf16 (Bt stored [N][K]): lane col N=lo; K chunks [hi*8..hi*8+7], [16+hi*8..+7]
static __device__ __forceinline__ v16bf load_frag_b(const __bf16* p) {
  v16bf f;
  ((bfx8*)&f)[0] = *(const bfx8*)(p);
  ((bfx8*)&f)[1] = *(const bfx8*)(p + 16);
  return f;
}

static __device__ __forceinline__ v8f wmma_bf16(v16bf a, v16bf b, v8f c) {
  return __builtin_amdgcn_wmma_f32_16x16x32_bf16(false, a, false, b, (short)0, c,
                                                 false, false);
}

// --- kernel 0: f32 -> bf16 conversion (bandwidth trivial) ------------------
__global__ __launch_bounds__(256) void cvt_bf16_kernel(
    const float* __restrict__ s, __bf16* __restrict__ d, int n8) {
  const int t = blockIdx.x * 256 + threadIdx.x;
  if (t < n8) {
    const float4* sp = (const float4*)(s + (size_t)t * 8);
    float4 a = sp[0], b = sp[1];
    bfx8 o;
    o[0] = (__bf16)a.x; o[1] = (__bf16)a.y; o[2] = (__bf16)a.z; o[3] = (__bf16)a.w;
    o[4] = (__bf16)b.x; o[5] = (__bf16)b.y; o[6] = (__bf16)b.z; o[7] = (__bf16)b.w;
    *(bfx8*)(d + (size_t)t * 8) = o;
  }
}

// --- shared bf16 GEMM mainloop: C[128x128] of A[M,K] x B[N,K]^T ------------
// Both operands bf16, K-contiguous rows. 8 waves as 4(M) x 2(N).
__device__ __forceinline__ void gemm_mainloop(const __bf16* __restrict__ A,
                                              const __bf16* __restrict__ B,
                                              int K, v8f (&acc)[2][4]) {
  __shared__ __align__(16) __bf16 As[2][BM * LDSS];
  __shared__ __align__(16) __bf16 Bs[2][BN * LDSS];

  const int tid  = threadIdx.x;
  const int lane = tid & 31;
  const int w    = tid >> 5;
  const int wm   = w >> 1;
  const int wn   = w & 1;
  const int lo   = lane & 15;
  const int hi   = (lane >> 4) & 1;
  const int lrow = tid >> 1;        // 0..127
  const int lcol = (tid & 1) * 16;  // 0 or 16

  const __bf16* ga = A + (size_t)lrow * K + lcol;
  const __bf16* gb = B + (size_t)lrow * K + lcol;
  const int NIT = K / BKK;

#if HAVE_ASYNC
  auto issue = [&](int n) {
    const int kb = n * BKK, buf = n & 1;
    __bf16* da = &As[buf][lrow * LDSS + lcol];
    __bf16* db = &Bs[buf][lrow * LDSS + lcol];
    async_cp16(ga + kb, da);
    async_cp16(ga + kb + 8, da + 8);
    async_cp16(gb + kb, db);
    async_cp16(gb + kb + 8, db + 8);
  };
  issue(0);
  for (int n = 0; n < NIT; ++n) {
    const int buf = n & 1;
    __builtin_amdgcn_s_wait_asynccnt(0);  // tile n landed (overlapped w/ n-1)
    __syncthreads();                      // all waves: tile n visible, n-1 free
    if (n + 1 < NIT) issue(n + 1);        // DMA next tile during compute
    v16bf af[2], bfr[4];
#pragma unroll
    for (int im = 0; im < 2; ++im)
      af[im] = load_frag_a(&As[buf][(wm * 32 + im * 16 + lo) * LDSS + hi * 8]);
#pragma unroll
    for (int jn = 0; jn < 4; ++jn)
      bfr[jn] = load_frag_b(&Bs[buf][(wn * 64 + jn * 16 + lo) * LDSS + hi * 8]);
#pragma unroll
    for (int im = 0; im < 2; ++im)
#pragma unroll
      for (int jn = 0; jn < 4; ++jn)
        acc[im][jn] = wmma_bf16(af[im], bfr[jn], acc[im][jn]);
  }
#else
  for (int n = 0; n < NIT; ++n) {
    const int kb = n * BKK, buf = n & 1;
    {
      __bf16* da = &As[buf][lrow * LDSS + lcol];
      __bf16* db = &Bs[buf][lrow * LDSS + lcol];
      ((bfx8*)da)[0] = *(const bfx8*)(ga + kb);
      ((bfx8*)da)[1] = *(const bfx8*)(ga + kb + 8);
      ((bfx8*)db)[0] = *(const bfx8*)(gb + kb);
      ((bfx8*)db)[1] = *(const bfx8*)(gb + kb + 8);
    }
    __syncthreads();
    v16bf af[2], bfr[4];
#pragma unroll
    for (int im = 0; im < 2; ++im)
      af[im] = load_frag_a(&As[buf][(wm * 32 + im * 16 + lo) * LDSS + hi * 8]);
#pragma unroll
    for (int jn = 0; jn < 4; ++jn)
      bfr[jn] = load_frag_b(&Bs[buf][(wn * 64 + jn * 16 + lo) * LDSS + hi * 8]);
#pragma unroll
    for (int im = 0; im < 2; ++im)
#pragma unroll
      for (int jn = 0; jn < 4; ++jn)
        acc[im][jn] = wmma_bf16(af[im], bfr[jn], acc[im][jn]);
    __syncthreads();
  }
#endif
}

// --- kernel 1: QKV projection + RoPE + scatter -----------------------------
__global__ __launch_bounds__(256) void qkv_rope_kernel(
    const __bf16* __restrict__ xb, const __bf16* __restrict__ Wqkvb,
    __bf16* __restrict__ q_ws, __bf16* __restrict__ k_ws,
    __bf16* __restrict__ vt_ws) {
  const int mBase = blockIdx.x * BM;  // token rows (b*L + l)
  const int nBase = blockIdx.y * BN;  // qkv output cols

  v8f acc[2][4] = {};
  gemm_mainloop(xb + (size_t)mBase * DMODEL, Wqkvb + (size_t)nBase * DMODEL,
                DMODEL, acc);

  const int tid = threadIdx.x, lane = tid & 31, w = tid >> 5;
  const int wm = w >> 1, wn = w & 1, lo = lane & 15, hi = (lane >> 4) & 1;
  const int colBase = nBase + wn * 64;  // 64-aligned -> exactly one head

  // wave-uniform region + head index
  const int region = (colBase < 2048) ? 0 : ((colBase < 3072) ? 1 : 2);
  const int hq = (colBase - (region == 0 ? 0 : (region == 1 ? 2048 : 3072))) >> 6;

  // inv_freq per lane (fixed across rows): 10000^(-i/32), ln(1e4)/32
  float freqv[2];
#pragma unroll
  for (int jp = 0; jp < 2; ++jp)
    freqv[jp] = __expf(-(float)(jp * 16 + lo) * 0.28782313662425572f);

#pragma unroll
  for (int im = 0; im < 2; ++im) {
#pragma unroll
    for (int r = 0; r < 8; ++r) {
      const int row = mBase + wm * 32 + im * 16 + hi * 8 + r;
      const int b = row >> 11;
      const int l = row & 2047;
      const float lf = (float)l;
#pragma unroll
      for (int jp = 0; jp < 2; ++jp) {  // pair frags (jp, jp+2): dims i, i+32
        const int i = jp * 16 + lo;
        float t1 = acc[im][jp][r];
        float t2 = acc[im][jp + 2][r];
        float o1 = t1, o2 = t2;
        if (region < 2) {  // RoPE for Q and K
          const float ang = lf * freqv[jp];
          const float c = __cosf(ang), s = __sinf(ang);
          o1 = t1 * c - t2 * s;
          o2 = t2 * c + t1 * s;
        }
        const __bf16 b1 = (__bf16)o1, b2 = (__bf16)o2;
        if (region == 0) {        // Q: [b,h,l,d]
          __bf16* dst =
              q_ws + (((size_t)b * N_HEADS + hq) * LSEQ + l) * HEAD_DIM;
          dst[i] = b1;
          dst[i + 32] = b2;
        } else if (region == 1) { // K: [b,kvh,l,d]
          __bf16* dst =
              k_ws + (((size_t)b * N_KV_HEADS + hq) * LSEQ + l) * HEAD_DIM;
          dst[i] = b1;
          dst[i + 32] = b2;
        } else {                  // V transposed: [b,kvh,d,l]
          __bf16* dst =
              vt_ws + (((size_t)b * N_KV_HEADS + hq) * HEAD_DIM) * LSEQ + l;
          dst[(size_t)i * LSEQ] = b1;
          dst[(size_t)(i + 32) * LSEQ] = b2;
        }
      }
    }
  }
}

// --- kernel 2: causal flash attention (64-row q block / WG, 4 waves) -------
__global__ __launch_bounds__(128) void flash_kernel(
    const __bf16* __restrict__ q_ws, const __bf16* __restrict__ k_ws,
    const __bf16* __restrict__ vt_ws, __bf16* __restrict__ attn_ws) {
  __shared__ __align__(16) __bf16 Ks[2][64 * LDK];  // Bt[n=kv][k=d]
  __shared__ __align__(16) __bf16 Vs[2][64 * LDK];  // Bt[n=d][k=kv]
  __shared__ __align__(16) __bf16 Ps[4 * 16 * LDK]; // per-wave P staging

  const int qblk = blockIdx.x;
  const int bh   = blockIdx.y;
  const int b = bh >> 5, h = bh & 31;
  const int kvh = h >> 2;

  const int tid = threadIdx.x, lane = tid & 31, w = tid >> 5;
  const int lo = lane & 15, hi = (lane >> 4) & 1;
  const int q0 = qblk * 64 + w * 16;

  const __bf16* qB = q_ws + (((size_t)b * N_HEADS + h) * LSEQ) * HEAD_DIM;
  const __bf16* kB = k_ws + (((size_t)b * N_KV_HEADS + kvh) * LSEQ) * HEAD_DIM;
  const __bf16* vB = vt_ws + (((size_t)b * N_KV_HEADS + kvh) * HEAD_DIM) * LSEQ;

  v16bf qf[2];
#pragma unroll
  for (int kc = 0; kc < 2; ++kc)
    qf[kc] = load_frag_a(qB + (size_t)(q0 + lo) * HEAD_DIM + kc * 32 + hi * 8);

  v8f   of[4] = {};
  float mrow[8], lsum[8];
#pragma unroll
  for (int r = 0; r < 8; ++r) { mrow[r] = NEG_BIG; lsum[r] = 0.0f; }

  __bf16* Pw = Ps + w * 16 * LDK;
  const int nblk = qblk + 1;
  const int rrow = tid >> 1, cb = (tid & 1) * 32;

#if HAVE_ASYNC
  auto issueKV = [&](int n) {
    const int kb = n * 64, buf = n & 1;
    const __bf16* ksrc = kB + (size_t)(kb + rrow) * HEAD_DIM + cb;
    const __bf16* vsrc = vB + (size_t)rrow * LSEQ + kb + cb;
    __bf16* kdst = &Ks[buf][rrow * LDK + cb];
    __bf16* vdst = &Vs[buf][rrow * LDK + cb];
#pragma unroll
    for (int c = 0; c < 4; ++c) {
      async_cp16(ksrc + c * 8, kdst + c * 8);
      async_cp16(vsrc + c * 8, vdst + c * 8);
    }
  };
  issueKV(0);
#endif

  for (int n = 0; n < nblk; ++n) {
    const int kb = n * 64;
    const int buf = n & 1;
#if HAVE_ASYNC
    __builtin_amdgcn_s_wait_asynccnt(0);
    __syncthreads();
    if (n + 1 < nblk) issueKV(n + 1);
#else
    {
      const __bf16* ksrc = kB + (size_t)(kb + rrow) * HEAD_DIM + cb;
      const __bf16* vsrc = vB + (size_t)rrow * LSEQ + kb + cb;
      __bf16* kdst = &Ks[buf][rrow * LDK + cb];
      __bf16* vdst = &Vs[buf][rrow * LDK + cb];
#pragma unroll
      for (int c = 0; c < 4; ++c) {
        *(bfx8*)(kdst + c * 8) = *(const bfx8*)(ksrc + c * 8);
        *(bfx8*)(vdst + c * 8) = *(const bfx8*)(vsrc + c * 8);
      }
    }
    __syncthreads();
#endif

    if (kb <= q0 + 15) {  // wave-uniform: skip fully-masked diagonal overshoot
      v8f sf[4];
#pragma unroll
      for (int j = 0; j < 4; ++j) {
        v8f z = {};
#pragma unroll
        for (int kc = 0; kc < 2; ++kc) {
          v16bf bk =
              load_frag_b(&Ks[buf][(j * 16 + lo) * LDK + kc * 32 + hi * 8]);
          z = wmma_bf16(qf[kc], bk, z);
        }
        sf[j] = z;
      }

      float alpha[8];
#pragma unroll
      for (int r = 0; r < 8; ++r) {
        const int qrow = q0 + hi * 8 + r;
        float mx = mrow[r];
#pragma unroll
        for (int j = 0; j < 4; ++j) {
          const int kcol = kb + j * 16 + lo;
          float s = sf[j][r] * SCALE_C;
          s = (kcol <= qrow) ? s : NEG_BIG;
          sf[j][r] = s;
          mx = fmaxf(mx, s);
        }
#pragma unroll
        for (int off = 1; off < 16; off <<= 1)
          mx = fmaxf(mx, __shfl_xor(mx, off, 32));
        alpha[r] = __expf(mrow[r] - mx);
        mrow[r]  = mx;
        float sum = 0.0f;
#pragma unroll
        for (int j = 0; j < 4; ++j) {
          float p = __expf(sf[j][r] - mx);
          sf[j][r] = p;
          sum += p;
        }
#pragma unroll
        for (int off = 1; off < 16; off <<= 1)
          sum += __shfl_xor(sum, off, 32);
        lsum[r] = lsum[r] * alpha[r] + sum;
      }

      // P: C/D layout -> A layout via per-wave LDS pad (same-wave DS in-order)
#pragma unroll
      for (int j = 0; j < 4; ++j)
#pragma unroll
        for (int r = 0; r < 8; ++r)
          Pw[(hi * 8 + r) * LDK + j * 16 + lo] = (__bf16)sf[j][r];

#pragma unroll
      for (int jd = 0; jd < 4; ++jd)
#pragma unroll
        for (int r = 0; r < 8; ++r)
          of[jd][r] *= alpha[r];

#pragma unroll
      for (int kc = 0; kc < 2; ++kc) {
        v16bf aP = load_frag_a(Pw + lo * LDK + kc * 32 + hi * 8);
#pragma unroll
        for (int jd = 0; jd < 4; ++jd) {
          v16bf bV =
              load_frag_b(&Vs[buf][(jd * 16 + lo) * LDK + kc * 32 + hi * 8]);
          of[jd] = wmma_bf16(aP, bV, of[jd]);
        }
      }
    }
#if !HAVE_ASYNC
    __syncthreads();
#endif
  }

  // finalize: O /= l, store bf16 rows for the O-projection
#pragma unroll
  for (int r = 0; r < 8; ++r) lsum[r] = 1.0f / lsum[r];
#pragma unroll
  for (int jd = 0; jd < 4; ++jd)
#pragma unroll
    for (int r = 0; r < 8; ++r) {
      const int qrow = q0 + hi * 8 + r;
      const size_t idx =
          ((size_t)(b * LSEQ + qrow)) * DMODEL + h * HEAD_DIM + jd * 16 + lo;
      attn_ws[idx] = (__bf16)(of[jd][r] * lsum[r]);
    }
}

// --- kernel 3: output projection  out = attn @ Wo^T ------------------------
__global__ __launch_bounds__(256) void oproj_kernel(
    const __bf16* __restrict__ attn, const __bf16* __restrict__ Wob,
    float* __restrict__ out) {
  const int mBase = blockIdx.x * BM;
  const int nBase = blockIdx.y * BN;

  v8f acc[2][4] = {};
  gemm_mainloop(attn + (size_t)mBase * DMODEL, Wob + (size_t)nBase * DMODEL,
                DMODEL, acc);

  const int tid = threadIdx.x, lane = tid & 31, w = tid >> 5;
  const int wm = w >> 1, wn = w & 1, lo = lane & 15, hi = (lane >> 4) & 1;
#pragma unroll
  for (int im = 0; im < 2; ++im)
#pragma unroll
    for (int jn = 0; jn < 4; ++jn)
#pragma unroll
      for (int r = 0; r < 8; ++r) {
        const int row = mBase + wm * 32 + im * 16 + hi * 8 + r;
        const int col = nBase + wn * 64 + jn * 16 + lo;
        out[(size_t)row * DMODEL + col] = acc[im][jn][r];
      }
}

// ---------------------------------------------------------------------------
extern "C" void kernel_launch(void* const* d_in, const int* in_sizes, int n_in,
                              void* d_out, int out_size, void* d_ws,
                              size_t ws_size, hipStream_t stream) {
  const float* x    = (const float*)d_in[0];  // (2, 2048, 2048)
  const float* Wqkv = (const float*)d_in[1];  // (3072, 2048)
  const float* Wo   = (const float*)d_in[2];  // (2048, 2048) = [d][h]
  float*       out  = (float*)d_out;          // (2, 2048, 2048)

  const size_t nX  = (size_t)2 * LSEQ * DMODEL;       // 8,388,608
  const size_t nWq = (size_t)OPSZ * DMODEL;           // 6,291,456
  const size_t nWo = (size_t)DMODEL * DMODEL;         // 4,194,304
  const size_t nQ  = (size_t)2 * N_HEADS * LSEQ * HEAD_DIM;
  const size_t nKV = (size_t)2 * N_KV_HEADS * LSEQ * HEAD_DIM;

  __bf16* xb    = (__bf16*)d_ws;
  __bf16* wqb   = xb + nX;
  __bf16* wob   = wqb + nWq;
  __bf16* q_ws  = wob + nWo;
  __bf16* k_ws  = q_ws + nQ;
  __bf16* vt_ws = k_ws + nKV;
  __bf16* a_ws  = vt_ws + nKV;

  // 0) pre-convert operands to bf16 (enables pure async-DMA GEMM staging)
  cvt_bf16_kernel<<<(int)(nX / 8 / 256), 256, 0, stream>>>(x, xb, (int)(nX / 8));
  cvt_bf16_kernel<<<(int)(nWq / 8 / 256), 256, 0, stream>>>(Wqkv, wqb,
                                                            (int)(nWq / 8));
  cvt_bf16_kernel<<<(int)(nWo / 8 / 256), 256, 0, stream>>>(Wo, wob,
                                                            (int)(nWo / 8));

  // 1) QKV projection + RoPE + scatter: M=4096, N=3072
  qkv_rope_kernel<<<dim3(4096 / BM, OPSZ / BN), 256, 0, stream>>>(
      xb, wqb, q_ws, k_ws, vt_ws);

  // 2) causal flash attention: 32 q-blocks x 64 (b,h) pairs
  flash_kernel<<<dim3(LSEQ / 64, 2 * N_HEADS), 128, 0, stream>>>(
      q_ws, k_ws, vt_ws, a_ws);

  // 3) output projection: M=4096, N=2048
  oproj_kernel<<<dim3(4096 / BM, DMODEL / BN), 256, 0, stream>>>(a_ws, wob, out);
}
